// ScaledDotProductAttention_63599875719555
// MI455X (gfx1250) — matrix operand order repaired and checked by reading, not compiled, not run on hardware
//
#include <hip/hip_runtime.h>
#include <hip/hip_bf16.h>

// Problem constants (match reference)
constexpr int CB = 2, CH = 8, CS = 2048, CD = 64;
constexpr float SCALE = 0.125f; // 1/sqrt(64)

typedef __attribute__((ext_vector_type(16))) _Float16 v16h;
typedef __attribute__((ext_vector_type(8)))  float    v8f;

// ---------------------------------------------------------------------------
// Fragment builders for 16-bit WMMA operands, using contiguous k = 16*hi + e in both A and B.
// A-matrix 16x32: lane L holds row m=L&15; hi=L>>4 selects kb = hi*16.
//   vgpr v in 0..7 -> K = kb + 2v + p (same k order as B)
// B-matrix 32x16: lane L holds col n=L&15; kb = hi*16; vgpr v -> K = kb+2v+p.
// Each 8-element half-fragment is contiguous in LDS -> lowers to ds_load_b128.
// ---------------------------------------------------------------------------
__device__ __forceinline__ v16h frag_a_rows(const _Float16* row, int hi, int koff) {
  v16h a;
  const int kb = hi * 16;
#pragma unroll
  for (int e = 0; e < 16; ++e) {
    const int v = e >> 1, p = e & 1;
    const int k = kb + 2 * v + p;
    a[e] = row[k + koff];
  }
  return a;
}

// B fragment when the operand matrix is stored row-major as [n][k]
// (QK^T case: B[k][n] = Krows[n][k])
__device__ __forceinline__ v16h frag_b_rows(const _Float16* row, int hi, int koff) {
  v16h b;
  const int kb = hi * 16;
#pragma unroll
  for (int e = 0; e < 16; ++e) {
    const int v = e >> 1, p = e & 1;
    const int k = kb + 2 * v + p;
    b[e] = row[k + koff];
  }
  return b;
}

// B fragment when the operand matrix is stored [k][n] (PV case: V tile)
__device__ __forceinline__ v16h frag_b_cols(const _Float16* base, int ld, int col, int hi) {
  v16h b;
  const int kb = hi * 16;
#pragma unroll
  for (int e = 0; e < 16; ++e) {
    const int v = e >> 1, p = e & 1;
    const int k = kb + 2 * v + p;
    b[e] = base[k * ld + col];
  }
  return b;
}

__device__ __forceinline__ v8f wmma_f16(v16h a, v16h b, v8f c) {
  return __builtin_amdgcn_wmma_f32_16x16x32_f16(false, a, false, b, (short)0, c,
                                                false, false);
}

// ---------------------------------------------------------------------------
// Kernel 1: masked scaled scores  S = mask ? -1e9 : (Q K^T)/8
// grid: (S/128, S/128, B*H), block: 256 threads (8 waves)
// Each wave owns a 16-row strip of the 128x128 tile; 8 column tiles of 16.
// ---------------------------------------------------------------------------
__global__ __launch_bounds__(256) void sdpa_scores(const float* __restrict__ Q,
                                                   const float* __restrict__ Kmat,
                                                   const unsigned char* __restrict__ mask,
                                                   float* __restrict__ attn) {
  constexpr int LDQ = 72; // f16 stride with padding
  __shared__ _Float16 Qs[128 * LDQ];
  __shared__ _Float16 Ks[128 * LDQ];

  const int bh = blockIdx.z;
  const int mt = blockIdx.y;
  const int nt = blockIdx.x;
  const int tid = threadIdx.x;

  const float* Qg = Q    + ((size_t)bh * CS + (size_t)mt * 128) * CD;
  const float* Kg = Kmat + ((size_t)bh * CS + (size_t)nt * 128) * CD;

  // cooperative load: 128x64 f32 -> f16 LDS (vectorized float4)
  for (int idx = tid; idx < 2048; idx += 256) {
    const int row = idx >> 4;
    const int c4  = (idx & 15) << 2;
    float4 q = *(const float4*)(Qg + row * CD + c4);
    Qs[row * LDQ + c4 + 0] = (_Float16)q.x;
    Qs[row * LDQ + c4 + 1] = (_Float16)q.y;
    Qs[row * LDQ + c4 + 2] = (_Float16)q.z;
    Qs[row * LDQ + c4 + 3] = (_Float16)q.w;
    float4 k = *(const float4*)(Kg + row * CD + c4);
    Ks[row * LDQ + c4 + 0] = (_Float16)k.x;
    Ks[row * LDQ + c4 + 1] = (_Float16)k.y;
    Ks[row * LDQ + c4 + 2] = (_Float16)k.z;
    Ks[row * LDQ + c4 + 3] = (_Float16)k.w;
  }
  __syncthreads();

  const int w    = tid >> 5;
  const int lane = tid & 31;
  const int lm   = lane & 15;
  const int hi   = lane >> 4;

  const _Float16* qrow = Qs + (w * 16 + lm) * LDQ;
  const v16h a0 = frag_a_rows(qrow, hi, 0);
  const v16h a1 = frag_a_rows(qrow, hi, 32);

  // Base element offset for this lane's output column within the 128x128 tile:
  // rows advance by S elements; per-tile column step is +16.
  const int    row0    = mt * 128 + w * 16 + hi * 8;   // first of 8 rows
  const size_t base    = ((size_t)bh * CS + row0) * (size_t)CS + (size_t)nt * 128 + lm;

#pragma unroll
  for (int t = 0; t < 8; ++t) {
    const _Float16* krow = Ks + (t * 16 + lm) * LDQ;
    const v16h b0 = frag_b_rows(krow, hi, 0);
    const v16h b1 = frag_b_rows(krow, hi, 32);
    v8f acc = {};
    acc = wmma_f16(a0, b0, acc);
    acc = wmma_f16(a1, b1, acc);

    const size_t off = base + t * 16;

    // Batch the 8 mask-byte loads first so they issue back-to-back (MLP),
    // instead of a load->wait->store dependent chain per row.
    unsigned char mk[8];
#pragma unroll
    for (int r = 0; r < 8; ++r) mk[r] = mask[off + (size_t)r * CS];

#pragma unroll
    for (int r = 0; r < 8; ++r) {
      float val = acc[r] * SCALE;
      if (mk[r]) val = -1e9f;
      attn[off + (size_t)r * CS] = val;
    }
  }
}

// ---------------------------------------------------------------------------
// Kernel 2: row softmax in place over attn.  grid: B*H*S blocks of 256.
// ---------------------------------------------------------------------------
__global__ __launch_bounds__(256) void sdpa_softmax(float* __restrict__ attn) {
  __shared__ float red[256];
  const int tid = threadIdx.x;
  float* p = attn + (size_t)blockIdx.x * CS;

  float vals[8];
  float m = -3.4e38f;
#pragma unroll
  for (int i = 0; i < 8; ++i) {
    vals[i] = p[tid + i * 256];
    m = fmaxf(m, vals[i]);
  }
  red[tid] = m;
  __syncthreads();
  for (int s = 128; s > 0; s >>= 1) {
    if (tid < s) red[tid] = fmaxf(red[tid], red[tid + s]);
    __syncthreads();
  }
  const float rmax = red[0];
  __syncthreads();

  float sum = 0.f;
#pragma unroll
  for (int i = 0; i < 8; ++i) {
    vals[i] = __expf(vals[i] - rmax);
    sum += vals[i];
  }
  red[tid] = sum;
  __syncthreads();
  for (int s = 128; s > 0; s >>= 1) {
    if (tid < s) red[tid] += red[tid + s];
    __syncthreads();
  }
  const float inv = 1.0f / red[0];
#pragma unroll
  for (int i = 0; i < 8; ++i) p[tid + i * 256] = vals[i] * inv;
}

// ---------------------------------------------------------------------------
// Kernel 3: context = attn x V.  grid: (S/128, B*H), block 256 (8 waves).
// Workgroup computes 128 query rows x 64 dims; K loop in chunks of 32.
// ---------------------------------------------------------------------------
__global__ __launch_bounds__(256) void sdpa_context(const float* __restrict__ V,
                                                    const float* __restrict__ attn,
                                                    float* __restrict__ ctx) {
  constexpr int LDV = 72; // [32][64] tile, padded
  constexpr int LDP = 40; // [128][32] tile, padded
  __shared__ _Float16 Vs[32 * LDV];
  __shared__ _Float16 Ps[128 * LDP];

  const int bh = blockIdx.y;
  const int mt = blockIdx.x;
  const int tid = threadIdx.x;
  const int w    = tid >> 5;
  const int lane = tid & 31;
  const int lm   = lane & 15;
  const int hi   = lane >> 4;

  const float* Vg = V + (size_t)bh * CS * CD;
  const float* Pg = attn + ((size_t)bh * CS + (size_t)mt * 128) * (size_t)CS;

  v8f acc[4] = {};

#pragma unroll 2
  for (int kc = 0; kc < CS / 32; ++kc) {
    // V chunk [32][64] f32 -> f16 LDS
    for (int idx = tid; idx < 512; idx += 256) {
      const int row = idx >> 4;
      const int c4  = (idx & 15) << 2;
      float4 v = *(const float4*)(Vg + (size_t)(kc * 32 + row) * CD + c4);
      Vs[row * LDV + c4 + 0] = (_Float16)v.x;
      Vs[row * LDV + c4 + 1] = (_Float16)v.y;
      Vs[row * LDV + c4 + 2] = (_Float16)v.z;
      Vs[row * LDV + c4 + 3] = (_Float16)v.w;
    }
    // P chunk [128][32] f32 -> f16 LDS
    for (int idx = tid; idx < 1024; idx += 256) {
      const int row = idx >> 3;
      const int c4  = (idx & 7) << 2;
      float4 v = *(const float4*)(Pg + (size_t)row * CS + kc * 32 + c4);
      Ps[row * LDP + c4 + 0] = (_Float16)v.x;
      Ps[row * LDP + c4 + 1] = (_Float16)v.y;
      Ps[row * LDP + c4 + 2] = (_Float16)v.z;
      Ps[row * LDP + c4 + 3] = (_Float16)v.w;
    }
    __syncthreads();

    const v16h a = frag_a_rows(Ps + (w * 16 + lm) * LDP, hi, 0);
#pragma unroll
    for (int t = 0; t < 4; ++t) {
      const v16h b = frag_b_cols(Vs, LDV, t * 16 + lm, hi);
      acc[t] = wmma_f16(a, b, acc[t]);
    }
    __syncthreads();
  }

#pragma unroll
  for (int t = 0; t < 4; ++t) {
    const int n = t * 16 + lm;
#pragma unroll
    for (int r = 0; r < 8; ++r) {
      const int m = mt * 128 + w * 16 + r + hi * 8;
      ctx[((size_t)bh * CS + m) * CD + n] = acc[t][r];
    }
  }
}

extern "C" void kernel_launch(void* const* d_in, const int* in_sizes, int n_in,
                              void* d_out, int out_size, void* d_ws, size_t ws_size,
                              hipStream_t stream) {
  (void)in_sizes; (void)n_in; (void)out_size; (void)d_ws; (void)ws_size;
  const float* Q = (const float*)d_in[0];
  const float* K = (const float*)d_in[1];
  const float* V = (const float*)d_in[2];
  const unsigned char* mask = (const unsigned char*)d_in[3];
  // d_in[4] = d_k (scalar 64) -> folded into SCALE

  float* ctx  = (float*)d_out;                               // [B,H,S,D]
  float* attn = (float*)d_out + (size_t)CB * CH * CS * CD;   // [B,H,S,S]

  const dim3 blk(256);
  sdpa_scores<<<dim3(CS / 128, CS / 128, CB * CH), blk, 0, stream>>>(Q, K, mask, attn);
  sdpa_softmax<<<dim3(CB * CH * CS), blk, 0, stream>>>(attn);
  sdpa_context<<<dim3(CS / 128, CB * CH), blk, 0, stream>>>(V, attn, ctx);
}
